// WindowZigZagAttention_70342974374158
// MI455X (gfx1250) — compile-verified, matrix operand order checked by reading, not run on hardware
//
#include <hip/hip_runtime.h>
#include <hip/hip_bf16.h>
#include <cmath>

typedef __attribute__((ext_vector_type(16))) _Float16 v16h;
typedef __attribute__((ext_vector_type(8)))  _Float16 v8h;
typedef __attribute__((ext_vector_type(8)))  float    v8f;

#define NTHREADS 256
#define WS_WQKVT_OFF   0          // 384*128 f16  = 98304 B
#define WS_WPROJT_OFF  98304      // 128*128 f16  = 32768 B
#define WS_BIAS_OFF    131072     // 4*64*64 f32  = 65536 B

union Frag16 { v16h f; v8h h[2]; };

__device__ __forceinline__ v16h load_ab(const _Float16* p) {
  Frag16 f;
  f.h[0] = *(const v8h*)(p);
  f.h[1] = *(const v8h*)(p + 16);
  return f.f;
}

__device__ __forceinline__ v8f wmma_f16(v16h a, v16h b, v8f c) {
  // D = A x B + C, 16x16x32 f16 -> f32
  return __builtin_amdgcn_wmma_f32_16x16x32_f16(false, a, false, b, (short)0, c,
                                                false, false);
}

// ---------------------------------------------------------------------------
// Prologue: transpose/convert weights to f16, materialize bias [H][64][64] f32
// ---------------------------------------------------------------------------
__global__ __launch_bounds__(256) void wzz_prep_kernel(
    const float* __restrict__ w_qkv, const float* __restrict__ w_proj,
    const float* __restrict__ bias_table, const int* __restrict__ rel_index,
    _Float16* __restrict__ wqkvT, _Float16* __restrict__ wprojT,
    float* __restrict__ bias_mat)
{
  int tid = blockIdx.x * blockDim.x + threadIdx.x;
  if (tid < 384 * 128) {                       // wqkvT[n][k] = w_qkv[k][n]
    int n = tid >> 7, k = tid & 127;
    wqkvT[n * 128 + k] = (_Float16)w_qkv[k * 384 + n];
    return;
  }
  int t2 = tid - 384 * 128;
  if (t2 < 128 * 128) {                        // wprojT[n][k] = w_proj[k][n]
    int n = t2 >> 7, k = t2 & 127;
    wprojT[n * 128 + k] = (_Float16)w_proj[k * 128 + n];
    return;
  }
  int t3 = t2 - 128 * 128;
  if (t3 < 4 * 64 * 64) {                      // bias_mat[h][n][m] (zero pad)
    int h = t3 >> 12, n = (t3 >> 6) & 63, m = t3 & 63;
    float v = 0.f;
    if (n < 49 && m < 49) v = bias_table[rel_index[n * 49 + m] * 4 + h];
    bias_mat[t3] = v;
  }
}

// ---------------------------------------------------------------------------
// Fused window attention: 1 block = 1 window (49 tokens, padded to 64)
// ---------------------------------------------------------------------------
__global__ __launch_bounds__(256) void wzz_attn_kernel(
    const float* __restrict__ x, const float* __restrict__ b_qkv,
    const float* __restrict__ b_proj, const _Float16* __restrict__ wqkvT,
    const _Float16* __restrict__ wprojT, const float* __restrict__ bias_mat,
    float* __restrict__ out)
{
  // LDS: 17408 + 5120 + 5120 + 4608 + 17408 + 5120 = 54784 bytes
  __shared__ __align__(16) _Float16 sh_x[64 * 136];   // x tile f16
  __shared__ __align__(16) _Float16 sh_q[64 * 40];    // Q*scale (this head)
  __shared__ __align__(16) _Float16 sh_k[64 * 40];    // K (this head)
  __shared__ __align__(16) _Float16 sh_vt[32 * 72];   // V transposed [d][m]
  __shared__ __align__(16) float    sh_sp[64 * 68];   // S f32, aliased P f16
  __shared__ __align__(16) _Float16 sh_o[64 * 40];    // O (this head)
  _Float16* sh_p = (_Float16*)sh_sp;                  // P view: [64][136] f16

  const int tid  = threadIdx.x;
  const int wave = tid >> 5;
  const int lane = tid & 31;
  const int r    = lane & 15;   // A row / B col / C col within tile
  const int hi   = lane >> 4;   // half-select per ISA fragment layout
  const int b    = blockIdx.x;
  const float scale = 0.17677669529663687f;  // 1/sqrt(32)

  if (tid == 0) {
    __builtin_prefetch(wqkvT, 0, 0);
    __builtin_prefetch(wprojT, 0, 0);
  }

  // ---- Phase 1: stage x[b] (49x128 f32) into LDS as f16, zero pad rows ----
  const float* xg = x + (size_t)b * (49 * 128);
  for (int i = tid; i < 49 * 32; i += NTHREADS) {
    int row = i >> 5, c4 = (i & 31) << 2;
    float4 f = *(const float4*)(xg + row * 128 + c4);
    _Float16* d = sh_x + row * 136 + c4;
    d[0] = (_Float16)f.x; d[1] = (_Float16)f.y;
    d[2] = (_Float16)f.z; d[3] = (_Float16)f.w;
  }
  for (int i = tid; i < 15 * 128; i += NTHREADS) {
    int row = 49 + (i >> 7), col = i & 127;
    sh_x[row * 136 + col] = (_Float16)0.f;
  }
  __syncthreads();

  // Projection accumulators live in registers across the head loop.
  v8f cproj[4];
#pragma unroll
  for (int i = 0; i < 4; ++i)
    cproj[i] = (v8f){0.f, 0.f, 0.f, 0.f, 0.f, 0.f, 0.f, 0.f};

  for (int h = 0; h < 4; ++h) {
    // ---- QKV GEMM for this head: 24 tiles (4 M x 6 cols of 16), K=128 ----
    for (int t = wave; t < 24; t += 8) {
      int mt = t & 3, pt = t >> 2;
      int sec = pt >> 1, sub = pt & 1;           // sec: 0=Q 1=K 2=V
      int ncol = sec * 128 + h * 32 + sub * 16;  // column in full qkv space
      int m0 = mt << 4;
      v8f c = (v8f){0.f, 0.f, 0.f, 0.f, 0.f, 0.f, 0.f, 0.f};
#pragma unroll
      for (int kc = 0; kc < 128; kc += 32) {
        v16h a  = load_ab(sh_x + (m0 + r) * 136 + kc + hi * 8);
        v16h bb = load_ab(wqkvT + (ncol + r) * 128 + kc + hi * 8);
        c = wmma_f16(a, bb, c);
      }
      float bq = b_qkv[ncol + r];
      int cl = (sub << 4) + r;                   // 0..31 within head
#pragma unroll
      for (int v = 0; v < 8; ++v) {
        int row = m0 + (hi << 3) + v;
        float val = (row < 49) ? (c[v] + bq) : 0.f;
        if (sec == 0)       sh_q[row * 40 + cl]  = (_Float16)(val * scale);
        else if (sec == 1)  sh_k[row * 40 + cl]  = (_Float16)val;
        else                sh_vt[cl * 72 + row] = (_Float16)val;
      }
    }
    __syncthreads();

    // ---- S = (scale*Q) K^T + bias : 16 tiles, single K=32 WMMA each ----
    for (int t = wave; t < 16; t += 8) {
      int mt = t & 3, nt = t >> 2;
      int m0 = mt << 4, n0 = nt << 4;
      v8f c;
#pragma unroll
      for (int v = 0; v < 8; ++v)
        c[v] = bias_mat[(h * 64 + m0 + (hi << 3) + v) * 64 + n0 + r];
      v16h a  = load_ab(sh_q + (m0 + r) * 40 + hi * 8);
      v16h bb = load_ab(sh_k + (n0 + r) * 40 + hi * 8);
      c = wmma_f16(a, bb, c);
#pragma unroll
      for (int v = 0; v < 8; ++v)
        sh_sp[(m0 + (hi << 3) + v) * 68 + n0 + r] = c[v];
    }
    __syncthreads();

    // ---- Row softmax over 49 valid cols; write P (f16) in place ----
    if (tid < 64) {
      float* srow = sh_sp + tid * 68;
      _Float16* prow = sh_p + tid * 136;
      float m = -1e30f;
      for (int c = 0; c < 49; ++c) m = fmaxf(m, srow[c]);
      float sum = 0.f;
      for (int c = 0; c < 49; ++c) {   // prow[c] clobbers srow[c/2]: c/2<=c, safe
        float e = __expf(srow[c] - m);
        sum += e;
        prow[c] = (_Float16)e;
      }
      float inv = 1.f / sum;
      for (int c = 0; c < 49; ++c) prow[c] = (_Float16)((float)prow[c] * inv);
      for (int c = 49; c < 64; ++c) prow[c] = (_Float16)0.f;
    }
    __syncthreads();

    // ---- O = P V : 8 tiles (4 M x 2 of 16 cols), K=64 -> 2 WMMAs ----
    {
      int mt = wave & 3, dt = wave >> 2;
      int m0 = mt << 4;
      v8f c = (v8f){0.f, 0.f, 0.f, 0.f, 0.f, 0.f, 0.f, 0.f};
#pragma unroll
      for (int kc = 0; kc < 64; kc += 32) {
        v16h a  = load_ab(sh_p + (m0 + r) * 136 + kc + hi * 8);
        v16h bb = load_ab(sh_vt + (dt * 16 + r) * 72 + kc + hi * 8);
        c = wmma_f16(a, bb, c);
      }
#pragma unroll
      for (int v = 0; v < 8; ++v)
        sh_o[(m0 + (hi << 3) + v) * 40 + dt * 16 + r] = (_Float16)c[v];
    }
    __syncthreads();

    // ---- Partial projection: out += O_h @ w_projT[:, k=h*32..h*32+31] ----
    // wave owns n-tile = wave (cols wave*16..+15), all 4 m-tiles, K=32.
#pragma unroll
    for (int i = 0; i < 4; ++i) {
      int m0 = i << 4, n0 = wave << 4;
      v16h a  = load_ab(sh_o + (m0 + r) * 40 + hi * 8);
      v16h bb = load_ab(wprojT + (n0 + r) * 128 + h * 32 + hi * 8);
      cproj[i] = wmma_f16(a, bb, cproj[i]);
    }
    __syncthreads();
  }

  // ---- Final store: add b_proj, write valid rows only ----
  {
    int n0 = wave << 4;
    float bp = b_proj[n0 + r];
#pragma unroll
    for (int i = 0; i < 4; ++i) {
      int m0 = i << 4;
#pragma unroll
      for (int v = 0; v < 8; ++v) {
        int row = m0 + (hi << 3) + v;
        if (row < 49)
          out[((size_t)b * 49 + row) * 128 + n0 + r] = cproj[i][v] + bp;
      }
    }
  }
}

// ---------------------------------------------------------------------------
extern "C" void kernel_launch(void* const* d_in, const int* in_sizes, int n_in,
                              void* d_out, int out_size, void* d_ws, size_t ws_size,
                              hipStream_t stream) {
  const float* x          = (const float*)d_in[0];
  const float* w_qkv      = (const float*)d_in[1];
  const float* b_qkv      = (const float*)d_in[2];
  const float* bias_table = (const float*)d_in[3];
  const float* w_proj     = (const float*)d_in[4];
  const float* b_proj     = (const float*)d_in[5];
  const int*   rel_index  = (const int*)d_in[6];

  char* ws = (char*)d_ws;
  _Float16* wqkvT  = (_Float16*)(ws + WS_WQKVT_OFF);
  _Float16* wprojT = (_Float16*)(ws + WS_WPROJT_OFF);
  float*    biasm  = (float*)(ws + WS_BIAS_OFF);

  // 384*128 + 128*128 + 4*64*64 = 81920 work items
  wzz_prep_kernel<<<320, 256, 0, stream>>>(w_qkv, w_proj, bias_table, rel_index,
                                           wqkvT, wprojT, biasm);
  wzz_attn_kernel<<<4096, 256, 0, stream>>>(x, b_qkv, b_proj, wqkvT, wprojT,
                                            biasm, (float*)d_out);
}
